// BiDAF_13511967113426
// MI455X (gfx1250) — matrix operand order repaired and checked by reading, not compiled, not run on hardware
//
#include <hip/hip_runtime.h>
#include <hip/hip_bf16.h>
#include <math.h>

#define H 100
#define CD 8
#define CC 100
#define CW 5
#define WD 100
#define BB 64
#define LCC 512
#define LQQ 64
#define WMAX 16

typedef __attribute__((ext_vector_type(16))) __bf16 bf16x16;
typedef __attribute__((ext_vector_type(8)))  __bf16 bf16x8;
typedef __attribute__((ext_vector_type(8)))  float  f32x8;

// ---------------------------------------------------------------------------
// Pack a row-major (N,K) fp32 matrix into WMMA B-fragment layout:
//   P[((kt*nnt + nt)*32 + lane)*16 + e] = bf16( W[(nt*16+(lane&15))*K + kt*32 + kk] )
// (kk = (e>>3)*16 + (lane>>4)*8 + (e&7): lane%16 <-> N index, element <-> K index; zero padded).
// ---------------------------------------------------------------------------
__global__ void pack_b_kernel(const float* __restrict__ W, long long bsW,
                              __bf16* __restrict__ P, long long bsP,
                              int N, int K, int nkt, int nnt)
{
    long long total = (long long)nkt * nnt * 512;
    long long idx = (long long)blockIdx.x * blockDim.x + threadIdx.x;
    if (idx >= total) return;
    int e    = (int)(idx & 15);
    int lane = (int)((idx >> 4) & 31);
    long long t2 = idx >> 9;
    int nt = (int)(t2 % nnt);
    int kt = (int)(t2 / nnt);
    int n = nt * 16 + (lane & 15);
    int k = kt * 32 + ((e >> 3) << 4) + ((lane >> 4) << 3) + (e & 7);
    float v = (n < N && k < K) ? W[(size_t)blockIdx.y * bsW + (size_t)n * K + k] : 0.f;
    P[(size_t)blockIdx.y * bsP + idx] = (__bf16)v;
}

// ---------------------------------------------------------------------------
// WMMA bf16 GEMM: C[M,N] = act( A[M,Kp](bf16, zero-padded) @ B + bias )
// B pre-packed in fragment layout. One wave per 16x16 C tile. Inner loop is
// 2x b128 A loads + 1x b256 B load + v_wmma, no predication.
// ---------------------------------------------------------------------------
__global__ void __launch_bounds__(128)
gemm_bf16_wmma(const __bf16* __restrict__ A, long long bsA, int lda,
               const __bf16* __restrict__ Bp, long long bsB,
               const float* __restrict__ bias,
               float* __restrict__ C, long long bsC, int ldc,
               int M, int N, int act)
{
    int batch = blockIdx.y;
    A  += (long long)batch * bsA;
    Bp += (long long)batch * bsB;
    C  += (long long)batch * bsC;

    int nnt    = (N + 15) >> 4;
    int tilesM = M >> 4;                 // M always a multiple of 16 here
    int nkt    = lda >> 5;               // lda is K padded to 32
    int wave   = threadIdx.x >> 5;
    int tile   = blockIdx.x * 4 + wave;
    if (tile >= tilesM * nnt) return;    // wave-uniform: EXEC stays all-1s

    int tm = tile / nnt, tn = tile % nnt;
    int lane = threadIdx.x & 31;
    int m0 = tm << 4, n0 = tn << 4;

    f32x8 acc = {0.f, 0.f, 0.f, 0.f, 0.f, 0.f, 0.f, 0.f};
    // A: lane m<16 holds K 0-7 / 16-23, lane 16+m holds K 8-15 / 24-31
    const __bf16* arow = A + (size_t)(m0 + (lane & 15)) * lda + ((lane >> 4) << 3);
    const __bf16* bptr = Bp + ((size_t)tn * 32 + lane) * 16;
    size_t bstep = (size_t)nnt * 512;

    #pragma unroll 2
    for (int kt = 0; kt < nkt; ++kt) {
        const __bf16* ak = arow + (kt << 5);
        bf16x8 alo = *(const bf16x8*)(ak);
        bf16x8 ahi = *(const bf16x8*)(ak + 16);
        bf16x16 a = __builtin_shufflevector(alo, ahi,
                        0,1,2,3,4,5,6,7,8,9,10,11,12,13,14,15);
        bf16x16 b = *(const bf16x16*)(bptr + (size_t)kt * bstep);
        acc = __builtin_amdgcn_wmma_f32_16x16x32_bf16(
                  false, a, false, b, (short)0, acc, false, false);
    }

    // C layout: acc[v] in lane l -> row m0 + (l>=16 ? 8 : 0) + v, col n0 + (l&15)
    int cn = n0 + (lane & 15);
    if (cn < N) {
        int cm = m0 + ((lane >> 4) << 3);
        float bv = bias ? bias[cn] : 0.f;
        #pragma unroll
        for (int v = 0; v < 8; ++v) {
            float x = acc[v] + bv;
            if (act == 1) x = fmaxf(x, 0.f);
            else if (act == 2) x = 1.f / (1.f + __expf(-x));
            C[(size_t)(cm + v) * ldc + cn] = x;
        }
    }
}

// ---------------------------------------------------------------------------
// Char CNN: one block per word; LDS-staged char embeddings; conv+relu+maxpool.
// ---------------------------------------------------------------------------
__global__ void __launch_bounds__(128)
charcnn_kernel(const int* __restrict__ ids, const float* __restrict__ emb,
               const float* __restrict__ convw, const float* __restrict__ convb,
               float* __restrict__ outF, __bf16* __restrict__ outBF,
               int ldf, int ldbf)
{
    int n = blockIdx.x;
    int t = threadIdx.x;
    __shared__ float e[WMAX * CD];
    if (t < WMAX * CD) {
        int p = t / CD, d = t % CD;
        int id = ids[(size_t)n * WMAX + p];
        e[t] = emb[(size_t)id * CD + d];
    }
    __syncthreads();
    if (t < CC) {
        float bo = convb[t];
        float best = -1e30f;
        for (int p = 0; p <= WMAX - CW; ++p) {
            float s = bo;
            #pragma unroll
            for (int w = 0; w < CW; ++w)
                #pragma unroll
                for (int d = 0; d < CD; ++d)
                    s += e[(p + w) * CD + d] * convw[(t * CD + d) * CW + w];
            best = fmaxf(best, s);
        }
        float y = fmaxf(best, 0.f);
        outF[(size_t)n * ldf + t] = y;
        outBF[(size_t)n * ldbf + t] = (__bf16)y;
    }
}

__global__ void word_emb_kernel(const int* __restrict__ wid,
                                const float* __restrict__ wemb,
                                float* __restrict__ outF, __bf16* __restrict__ outBF,
                                int ldf, int ldbf, long long total)
{
    long long idx = (long long)blockIdx.x * blockDim.x + threadIdx.x;
    if (idx >= total) return;
    long long row = idx / WD; int d = (int)(idx % WD);
    float v = wemb[(size_t)wid[row] * WD + d];
    outF[(size_t)row * ldf + CC + d] = v;
    outBF[(size_t)row * ldbf + CC + d] = (__bf16)v;
}

__global__ void pad_bf_kernel(__bf16* __restrict__ buf, int ld, int K, long long rows)
{
    int padw = ld - K;
    long long idx = (long long)blockIdx.x * blockDim.x + threadIdx.x;
    if (idx >= rows * padw) return;
    long long row = idx / padw; int d = K + (int)(idx % padw);
    buf[(size_t)row * ld + d] = (__bf16)0.f;
}

__global__ void highway_combine(float* __restrict__ xf, __bf16* __restrict__ xbf,
                                const float* __restrict__ hf, const float* __restrict__ gf,
                                int ldbf, long long total)
{
    long long idx = (long long)blockIdx.x * blockDim.x + threadIdx.x;
    if (idx >= total) return;
    long long row = idx / 200; int d = (int)(idx % 200);
    float g = gf[idx], h = hf[idx], x = xf[idx];
    float y = g * h + (1.f - g) * x;
    xf[idx] = y;
    xbf[(size_t)row * ldbf + d] = (__bf16)y;
}

// ---------------------------------------------------------------------------
// LSTM recurrence: one 512-thread block per (batch row, direction).
// ---------------------------------------------------------------------------
__global__ void __launch_bounds__(512)
lstm_scan(const float* __restrict__ gates0, const float* __restrict__ gates1,
          const float* __restrict__ Whh,     // (2, 4H, H)
          float* __restrict__ outF, __bf16* __restrict__ outBF,
          int T, int ldf, int ldbf)
{
    int b = blockIdx.x, dir = blockIdx.y;
    const float* gates = dir ? gates1 : gates0;
    const float* Wd = Whh + (size_t)dir * 4 * H * H;
    __shared__ float h[H], c[H], gsh[4 * H];
    int t = threadIdx.x;
    if (t < H) { h[t] = 0.f; c[t] = 0.f; }
    __syncthreads();
    for (int step = 0; step < T; ++step) {
        int tt = dir ? (T - 1 - step) : step;
        const float* grow = gates + ((size_t)b * T + tt) * (4 * H);
        if (t < 4 * H) {
            float acc = grow[t];
            const float* wr = Wd + (size_t)t * H;
            #pragma unroll 4
            for (int k = 0; k < H; ++k) acc += h[k] * wr[k];
            gsh[t] = acc;
        }
        __syncthreads();
        if (t < H) {
            float ig = 1.f / (1.f + __expf(-gsh[t]));
            float fg = 1.f / (1.f + __expf(-gsh[H + t]));
            float gg = tanhf(gsh[2 * H + t]);
            float og = 1.f / (1.f + __expf(-gsh[3 * H + t]));
            float cn = fg * c[t] + ig * gg;
            float hn = og * tanhf(cn);
            c[t] = cn; h[t] = hn;
            size_t o = (size_t)b * T + tt;
            outF[o * ldf + dir * H + t] = hn;
            if (outBF) outBF[o * ldbf + dir * H + t] = (__bf16)hn;
        }
        __syncthreads();
    }
}

__global__ void row_dot_kernel(const float* __restrict__ X, int ld,
                               const float* __restrict__ w, int K,
                               float* __restrict__ out, int M)
{
    int r = blockIdx.x * blockDim.x + threadIdx.x;
    if (r >= M) return;
    const float* x = X + (size_t)r * ld;
    float s = 0.f;
    for (int k = 0; k < K; ++k) s += x[k] * w[k];
    out[r] = s;
}

__global__ void scale_cq_kernel(const float* __restrict__ cf, const float* __restrict__ wcq,
                                __bf16* __restrict__ cqbf, int ldbf, long long total)
{
    long long idx = (long long)blockIdx.x * blockDim.x + threadIdx.x;
    if (idx >= total) return;
    long long row = idx / 200; int d = (int)(idx % 200);
    cqbf[(size_t)row * ldbf + d] = (__bf16)(cf[idx] * wcq[d]);
}

// softmax over j (LQ=64) + c2q contraction; one 64-thread block per (b, i)
__global__ void __launch_bounds__(64)
att_softmax_kernel(const float* __restrict__ S, const float* __restrict__ cw,
                   const float* __restrict__ qw, const float* __restrict__ attb,
                   const float* __restrict__ qf,
                   float* __restrict__ c2q, float* __restrict__ mrow)
{
    int b = blockIdx.y, i = blockIdx.x, j = threadIdx.x;
    __shared__ float a[LQQ];
    __shared__ float mx, sum;
    size_t rowi = (size_t)b * LCC + i;
    float s = S[rowi * LQQ + j] + cw[rowi] + qw[b * LQQ + j] + attb[0];
    a[j] = s;
    __syncthreads();
    if (j == 0) {
        float m = -1e30f;
        for (int u = 0; u < LQQ; ++u) m = fmaxf(m, a[u]);
        mx = m; mrow[rowi] = m;
    }
    __syncthreads();
    float e = __expf(s - mx);
    a[j] = e;
    __syncthreads();
    if (j == 0) {
        float t = 0.f;
        for (int u = 0; u < LQQ; ++u) t += a[u];
        sum = t;
    }
    __syncthreads();
    float inv = 1.f / sum;
    for (int d = j; d < 2 * H; d += LQQ) {
        float acc = 0.f;
        for (int u = 0; u < LQQ; ++u)
            acc += a[u] * qf[((size_t)b * LQQ + u) * (2 * H) + d];
        c2q[rowi * (2 * H) + d] = acc * inv;
    }
}

// softmax over i (LC=512) of max_j s, then q2c; one block per batch
__global__ void __launch_bounds__(256)
q2c_kernel(const float* __restrict__ mrow, const float* __restrict__ cf,
           float* __restrict__ q2c)
{
    int b = blockIdx.x, t = threadIdx.x;
    __shared__ float batt[LCC];
    __shared__ float mx, sum;
    if (t == 0) {
        float m = -1e30f;
        for (int u = 0; u < LCC; ++u) m = fmaxf(m, mrow[(size_t)b * LCC + u]);
        mx = m;
    }
    __syncthreads();
    for (int u = t; u < LCC; u += 256)
        batt[u] = __expf(mrow[(size_t)b * LCC + u] - mx);
    __syncthreads();
    if (t == 0) {
        float s = 0.f;
        for (int u = 0; u < LCC; ++u) s += batt[u];
        sum = s;
    }
    __syncthreads();
    float inv = 1.f / sum;
    for (int d = t; d < 2 * H; d += 256) {
        float acc = 0.f;
        for (int u = 0; u < LCC; ++u)
            acc += batt[u] * cf[((size_t)b * LCC + u) * (2 * H) + d];
        q2c[(size_t)b * (2 * H) + d] = acc * inv;
    }
}

__global__ void build_g_kernel(const float* __restrict__ cf, const float* __restrict__ c2q,
                               const float* __restrict__ q2c,
                               float* __restrict__ gf, __bf16* __restrict__ gbf,
                               long long total)
{
    long long idx = (long long)blockIdx.x * blockDim.x + threadIdx.x;
    if (idx >= total) return;
    long long row = idx / (8 * H); int d = (int)(idx % (8 * H));
    long long b = row / LCC;
    float v;
    if (d < 2 * H)        v = cf[row * (2 * H) + d];
    else if (d < 4 * H)   v = c2q[row * (2 * H) + d - 2 * H];
    else if (d < 6 * H)   v = cf[row * (2 * H) + d - 4 * H] * c2q[row * (2 * H) + d - 4 * H];
    else                  v = cf[row * (2 * H) + d - 6 * H] * q2c[b * (2 * H) + d - 6 * H];
    gf[idx] = v;
    gbf[idx] = (__bf16)v;
}

__global__ void p_dot_kernel(const float* __restrict__ gf, const float* __restrict__ mf,
                             const float* __restrict__ wg, const float* __restrict__ wm,
                             const float* __restrict__ pb, float* __restrict__ out, int M)
{
    int r = blockIdx.x * blockDim.x + threadIdx.x;
    if (r >= M) return;
    float s = pb[0];
    const float* g = gf + (size_t)r * (8 * H);
    const float* m = mf + (size_t)r * (2 * H);
    for (int k = 0; k < 8 * H; ++k) s += g[k] * wg[k];
    for (int k = 0; k < 2 * H; ++k) s += m[k] * wm[k];
    out[r] = s;
}

// ---------------------------------------------------------------------------
static void launch_pack(const float* W, long long bsW, __bf16* P, long long bsP,
                        int N, int K, int Kp, int nbatch, hipStream_t s)
{
    int nkt = Kp >> 5, nnt = (N + 15) >> 4;
    long long total = (long long)nkt * nnt * 512;
    dim3 grid((unsigned)((total + 255) / 256), nbatch);
    pack_b_kernel<<<grid, 256, 0, s>>>(W, bsW, P, bsP, N, K, nkt, nnt);
}

static void launch_gemm(const __bf16* A, long long bsA, int lda,
                        const __bf16* Bp, long long bsB, const float* bias,
                        float* C, long long bsC, int ldc,
                        int M, int N, int act, int nbatch, hipStream_t s)
{
    int tiles = (M >> 4) * ((N + 15) >> 4);
    dim3 grid((tiles + 3) / 4, nbatch);
    gemm_bf16_wmma<<<grid, 128, 0, s>>>(A, bsA, lda, Bp, bsB, bias, C, bsC, ldc, M, N, act);
}

extern "C" void kernel_launch(void* const* d_in, const int* in_sizes, int n_in,
                              void* d_out, int out_size, void* d_ws, size_t ws_size,
                              hipStream_t stream)
{
    const float* char_emb  = (const float*)d_in[0];
    const float* conv_w    = (const float*)d_in[1];
    const float* conv_b    = (const float*)d_in[2];
    const float* word_embw = (const float*)d_in[3];
    const float* hw_lin_w  = (const float*)d_in[4];
    const float* hw_lin_b  = (const float*)d_in[5];
    const float* hw_gate_w = (const float*)d_in[6];
    const float* hw_gate_b = (const float*)d_in[7];
    const float* ctx_Wih   = (const float*)d_in[8];
    const float* ctx_Whh   = (const float*)d_in[9];
    const float* ctx_b     = (const float*)d_in[10];
    const float* mod1_Wih  = (const float*)d_in[11];
    const float* mod1_Whh  = (const float*)d_in[12];
    const float* mod1_b    = (const float*)d_in[13];
    const float* mod2_Wih  = (const float*)d_in[14];
    const float* mod2_Whh  = (const float*)d_in[15];
    const float* mod2_b    = (const float*)d_in[16];
    const float* out_Wih   = (const float*)d_in[17];
    const float* out_Whh   = (const float*)d_in[18];
    const float* out_b     = (const float*)d_in[19];
    const float* att_w_c   = (const float*)d_in[20];
    const float* att_w_q   = (const float*)d_in[21];
    const float* att_w_cq  = (const float*)d_in[22];
    const float* att_b     = (const float*)d_in[23];
    const float* p1_w_g    = (const float*)d_in[24];
    const float* p1_w_m    = (const float*)d_in[25];
    const float* p1_b      = (const float*)d_in[26];
    const float* p2_w_g    = (const float*)d_in[27];
    const float* p2_w_m    = (const float*)d_in[28];
    const float* p2_b      = (const float*)d_in[29];
    const int*   c_char    = (const int*)d_in[30];
    const int*   q_char    = (const int*)d_in[31];
    const int*   c_word    = (const int*)d_in[32];
    const int*   q_word    = (const int*)d_in[33];

    const int Mc = BB * LCC;     // 32768
    const int Mq = BB * LQQ;     // 4096
    const int D2 = 2 * H;        // 200
    const int Kp2 = 224;         // 200 padded to 32
    const int D8 = 8 * H;        // 800

    // packed B fragment sizes (bf16 elements)
    const long long bsP_hw   = 7LL  * 13 * 512;   // N=200, Kp=224
    const long long bsP_ctx  = 7LL  * 25 * 512;   // N=400, Kp=224
    const long long bsP_mod1 = 25LL * 25 * 512;   // N=400, Kp=800
    const long long bsP_q    = 7LL  * 4  * 512;   // N=64,  Kp=224

    size_t off = 0;
    auto alloc = [&](size_t bytes) -> void* {
        void* p = (char*)d_ws + off;
        off += (bytes + 255) & ~(size_t)255;
        return p;
    };
    float*  xc_f    = (float*) alloc((size_t)Mc * D2 * 4);
    __bf16* xc_bf   = (__bf16*)alloc((size_t)Mc * Kp2 * 2);
    float*  xq_f    = (float*) alloc((size_t)Mq * D2 * 4);
    __bf16* xq_bf   = (__bf16*)alloc((size_t)Mq * Kp2 * 2);
    float*  tmpH    = (float*) alloc((size_t)Mc * D2 * 4);
    float*  tmpG    = (float*) alloc((size_t)Mc * D2 * 4);
    float*  gatesA  = (float*) alloc((size_t)Mc * 4 * H * 4);
    float*  gatesB  = (float*) alloc((size_t)Mc * 4 * H * 4);
    float*  cOut_f  = (float*) alloc((size_t)Mc * D2 * 4);
    __bf16* cOut_bf = (__bf16*)alloc((size_t)Mc * Kp2 * 2);
    float*  qOut_f  = (float*) alloc((size_t)Mq * D2 * 4);
    float*  cw      = (float*) alloc((size_t)Mc * 4);
    float*  qw      = (float*) alloc((size_t)Mq * 4);
    float*  mrow    = (float*) alloc((size_t)Mc * 4);
    __bf16* cq_bf   = (__bf16*)alloc((size_t)Mc * Kp2 * 2);
    float*  Smat    = (float*) alloc((size_t)Mc * LQQ * 4);
    float*  c2q     = (float*) alloc((size_t)Mc * D2 * 4);
    float*  q2c     = (float*) alloc((size_t)BB * D2 * 4);
    float*  g_f     = (float*) alloc((size_t)Mc * D8 * 4);
    __bf16* g_bf    = (__bf16*)alloc((size_t)Mc * D8 * 2);
    float*  m1_f    = (float*) alloc((size_t)Mc * D2 * 4);
    __bf16* m1_bf   = (__bf16*)alloc((size_t)Mc * Kp2 * 2);
    float*  m_f     = (float*) alloc((size_t)Mc * D2 * 4);
    __bf16* m_bf    = (__bf16*)alloc((size_t)Mc * Kp2 * 2);
    float*  m2_f    = (float*) alloc((size_t)Mc * D2 * 4);
    __bf16* pk_lin  = (__bf16*)alloc((size_t)bsP_hw   * 2 * 2);
    __bf16* pk_gate = (__bf16*)alloc((size_t)bsP_hw   * 2 * 2);
    __bf16* pk_ctx  = (__bf16*)alloc((size_t)bsP_ctx  * 2 * 2);
    __bf16* pk_mod1 = (__bf16*)alloc((size_t)bsP_mod1 * 2 * 2);
    __bf16* pk_mod2 = (__bf16*)alloc((size_t)bsP_ctx  * 2 * 2);
    __bf16* pk_out  = (__bf16*)alloc((size_t)bsP_ctx  * 2 * 2);
    __bf16* pk_q    = (__bf16*)alloc((size_t)bsP_q    * BB * 2);

    float* p1_out = (float*)d_out;
    float* p2_out = (float*)d_out + Mc;

    // ---- pack all weight B-operands into WMMA fragment layout ----
    launch_pack(hw_lin_w,  (long long)D2 * D2,  pk_lin,  bsP_hw,   D2,    D2, Kp2, 2, stream);
    launch_pack(hw_gate_w, (long long)D2 * D2,  pk_gate, bsP_hw,   D2,    D2, Kp2, 2, stream);
    launch_pack(ctx_Wih,   (long long)4*H * D2, pk_ctx,  bsP_ctx,  4*H,   D2, Kp2, 2, stream);
    launch_pack(mod1_Wih,  (long long)4*H * D8, pk_mod1, bsP_mod1, 4*H,   D8, D8,  2, stream);
    launch_pack(mod2_Wih,  (long long)4*H * D2, pk_mod2, bsP_ctx,  4*H,   D2, Kp2, 2, stream);
    launch_pack(out_Wih,   (long long)4*H * D2, pk_out,  bsP_ctx,  4*H,   D2, Kp2, 2, stream);

    // ---- zero the pad columns of every bf16 activation buffer ----
    {
        long long tot = (long long)Mc * (Kp2 - D2);
        int blk = 256;
        int grd  = (int)((tot + blk - 1) / blk);
        int grdq = (int)(((long long)Mq * (Kp2 - D2) + blk - 1) / blk);
        pad_bf_kernel<<<grd,  blk, 0, stream>>>(xc_bf,   Kp2, D2, Mc);
        pad_bf_kernel<<<grdq, blk, 0, stream>>>(xq_bf,   Kp2, D2, Mq);
        pad_bf_kernel<<<grd,  blk, 0, stream>>>(cOut_bf, Kp2, D2, Mc);
        pad_bf_kernel<<<grd,  blk, 0, stream>>>(cq_bf,   Kp2, D2, Mc);
        pad_bf_kernel<<<grd,  blk, 0, stream>>>(m1_bf,   Kp2, D2, Mc);
        pad_bf_kernel<<<grd,  blk, 0, stream>>>(m_bf,    Kp2, D2, Mc);
    }

    // ---- 1-2: char CNN + word embeddings ----
    charcnn_kernel<<<Mc, 128, 0, stream>>>(c_char, char_emb, conv_w, conv_b,
                                           xc_f, xc_bf, D2, Kp2);
    charcnn_kernel<<<Mq, 128, 0, stream>>>(q_char, char_emb, conv_w, conv_b,
                                           xq_f, xq_bf, D2, Kp2);
    {
        long long totc = (long long)Mc * WD, totq = (long long)Mq * WD;
        word_emb_kernel<<<(int)((totc + 255) / 256), 256, 0, stream>>>(
            c_word, word_embw, xc_f, xc_bf, D2, Kp2, totc);
        word_emb_kernel<<<(int)((totq + 255) / 256), 256, 0, stream>>>(
            q_word, word_embw, xq_f, xq_bf, D2, Kp2, totq);
    }

    // ---- 3: highway (2 layers), context then query ----
    for (int i = 0; i < 2; ++i) {
        const __bf16* lw = pk_lin  + (size_t)i * bsP_hw;
        const __bf16* gw = pk_gate + (size_t)i * bsP_hw;
        const float*  lb = hw_lin_b  + (size_t)i * D2;
        const float*  gb = hw_gate_b + (size_t)i * D2;
        launch_gemm(xc_bf, 0, Kp2, lw, 0, lb, tmpH, 0, D2, Mc, D2, 1, 1, stream);
        launch_gemm(xc_bf, 0, Kp2, gw, 0, gb, tmpG, 0, D2, Mc, D2, 2, 1, stream);
        long long tot = (long long)Mc * D2;
        highway_combine<<<(int)((tot + 255) / 256), 256, 0, stream>>>(
            xc_f, xc_bf, tmpH, tmpG, Kp2, tot);
        launch_gemm(xq_bf, 0, Kp2, lw, 0, lb, tmpH, 0, D2, Mq, D2, 1, 1, stream);
        launch_gemm(xq_bf, 0, Kp2, gw, 0, gb, tmpG, 0, D2, Mq, D2, 2, 1, stream);
        long long totq = (long long)Mq * D2;
        highway_combine<<<(int)((totq + 255) / 256), 256, 0, stream>>>(
            xq_f, xq_bf, tmpH, tmpG, Kp2, totq);
    }

    // ---- 4: contextual BiLSTM (shared weights) ----
    launch_gemm(xc_bf, 0, Kp2, pk_ctx,           0, ctx_b,       gatesA, 0, 4*H, Mc, 4*H, 0, 1, stream);
    launch_gemm(xc_bf, 0, Kp2, pk_ctx + bsP_ctx, 0, ctx_b + 4*H, gatesB, 0, 4*H, Mc, 4*H, 0, 1, stream);
    lstm_scan<<<dim3(BB, 2), 512, 0, stream>>>(gatesA, gatesB, ctx_Whh,
                                               cOut_f, cOut_bf, LCC, D2, Kp2);
    launch_gemm(xq_bf, 0, Kp2, pk_ctx,           0, ctx_b,       gatesA, 0, 4*H, Mq, 4*H, 0, 1, stream);
    launch_gemm(xq_bf, 0, Kp2, pk_ctx + bsP_ctx, 0, ctx_b + 4*H, gatesB, 0, 4*H, Mq, 4*H, 0, 1, stream);
    lstm_scan<<<dim3(BB, 2), 512, 0, stream>>>(gatesA, gatesB, ctx_Whh,
                                               qOut_f, ((__bf16*)0), LQQ, D2, Kp2);

    // ---- 5: trilinear attention flow ----
    row_dot_kernel<<<(Mc + 255) / 256, 256, 0, stream>>>(cOut_f, D2, att_w_c, D2, cw, Mc);
    row_dot_kernel<<<(Mq + 255) / 256, 256, 0, stream>>>(qOut_f, D2, att_w_q, D2, qw, Mq);
    {
        long long tot = (long long)Mc * D2;
        scale_cq_kernel<<<(int)((tot + 255) / 256), 256, 0, stream>>>(
            cOut_f, att_w_cq, cq_bf, Kp2, tot);
    }
    // pack q[b] (64 x 200) per batch into fragment layout, then batched GEMM
    launch_pack(qOut_f, (long long)LQQ * D2, pk_q, bsP_q, LQQ, D2, Kp2, BB, stream);
    launch_gemm(cq_bf, (long long)LCC * Kp2, Kp2,
                pk_q, bsP_q, (const float*)0,
                Smat, (long long)LCC * LQQ, LQQ,
                LCC, LQQ, 0, BB, stream);
    att_softmax_kernel<<<dim3(LCC, BB), 64, 0, stream>>>(Smat, cw, qw, att_b,
                                                         qOut_f, c2q, mrow);
    q2c_kernel<<<BB, 256, 0, stream>>>(mrow, cOut_f, q2c);
    {
        long long tot = (long long)Mc * D8;
        build_g_kernel<<<(int)((tot + 255) / 256), 256, 0, stream>>>(
            cOut_f, c2q, q2c, g_f, g_bf, tot);
    }

    // ---- 6: modeling BiLSTM stack ----
    launch_gemm(g_bf, 0, D8, pk_mod1,            0, mod1_b,       gatesA, 0, 4*H, Mc, 4*H, 0, 1, stream);
    launch_gemm(g_bf, 0, D8, pk_mod1 + bsP_mod1, 0, mod1_b + 4*H, gatesB, 0, 4*H, Mc, 4*H, 0, 1, stream);
    lstm_scan<<<dim3(BB, 2), 512, 0, stream>>>(gatesA, gatesB, mod1_Whh,
                                               m1_f, m1_bf, LCC, D2, Kp2);
    launch_gemm(m1_bf, 0, Kp2, pk_mod2,           0, mod2_b,       gatesA, 0, 4*H, Mc, 4*H, 0, 1, stream);
    launch_gemm(m1_bf, 0, Kp2, pk_mod2 + bsP_ctx, 0, mod2_b + 4*H, gatesB, 0, 4*H, Mc, 4*H, 0, 1, stream);
    lstm_scan<<<dim3(BB, 2), 512, 0, stream>>>(gatesA, gatesB, mod2_Whh,
                                               m_f, m_bf, LCC, D2, Kp2);

    // ---- 7: output layer ----
    p_dot_kernel<<<(Mc + 255) / 256, 256, 0, stream>>>(g_f, m_f, p1_w_g, p1_w_m,
                                                       p1_b, p1_out, Mc);
    launch_gemm(m_bf, 0, Kp2, pk_out,           0, out_b,        gatesA, 0, 4*H, Mc, 4*H, 0, 1, stream);
    launch_gemm(m_bf, 0, Kp2, pk_out + bsP_ctx, 0, out_b + 4*H,  gatesB, 0, 4*H, Mc, 4*H, 0, 1, stream);
    lstm_scan<<<dim3(BB, 2), 512, 0, stream>>>(gatesA, gatesB, out_Whh,
                                               m2_f, ((__bf16*)0), LCC, D2, Kp2);
    p_dot_kernel<<<(Mc + 255) / 256, 256, 0, stream>>>(g_f, m2_f, p2_w_g, p2_w_m,
                                                       p2_b, p2_out, Mc);

    (void)in_sizes; (void)n_in; (void)out_size; (void)ws_size;
}
